// LegacySeqFirstSelfAttention_modified_7679401525984
// MI455X (gfx1250) — matrix operand order repaired and checked by reading, not compiled, not run on hardware
//
#include <hip/hip_runtime.h>

typedef __bf16 bf16_t;
typedef __attribute__((ext_vector_type(16))) __bf16 v16bf;
typedef __attribute__((ext_vector_type(8)))  __bf16 v8bf;
typedef __attribute__((ext_vector_type(8)))  float  v8f;

__device__ __forceinline__ bf16_t f2bf(float f) {
    unsigned u = __builtin_bit_cast(unsigned, f);
    unsigned r = (u + 0x7FFFu + ((u >> 16) & 1u)) >> 16;
    return __builtin_bit_cast(bf16_t, (unsigned short)r);
}
__device__ __forceinline__ float bf2f(bf16_t b) {
    unsigned short s = __builtin_bit_cast(unsigned short, b);
    return __builtin_bit_cast(float, (unsigned)s << 16);
}
__device__ __forceinline__ float polysig(float x) {
    float x2 = x * x;
    return 0.5f + x * (0.25f + x2 * (-2.0833333333e-2f + x2 * 2.0833333333e-3f));
}

// ---------------- conversion kernels ----------------
__global__ void cvt_hidden(const float* __restrict__ H, bf16_t* __restrict__ Hb, int n) {
    int i = blockIdx.x * blockDim.x + threadIdx.x;
    if (i < n) Hb[i] = f2bf(H[i]);
}
// W [1024][3072] f32 -> Wt [3072][1024] bf16
__global__ void cvt_weight_t(const float* __restrict__ W, bf16_t* __restrict__ Wt) {
    int i = blockIdx.x * blockDim.x + threadIdx.x;  // < 1024*3072
    int r = i / 3072, c = i % 3072;
    Wt[(size_t)c * 1024 + r] = f2bf(W[i]);
}

// ---------------- QKV block-diagonal GEMM (16x64 strip per wave) ----------------
// out[s, col] = sum_c Hb[s,b, j*256+c] * W[b*256+c, col], j = col/768
// Q,K -> QK [b][s][2048] (head g: q at g*128+d, k at g*128+64+d); V -> Vt [b][head][h=64][t=1024]
__global__ __launch_bounds__(128)
void qkv_gemm(const bf16_t* __restrict__ Hb, const bf16_t* __restrict__ Wt,
              bf16_t* __restrict__ QK, bf16_t* __restrict__ Vt) {
    const int lane = threadIdx.x & 31;
    const int wave = threadIdx.x >> 5;
    const int lhi = lane >> 4, llo = lane & 15;
    const int sTile = blockIdx.x;                 // 0..63
    const int grp = blockIdx.y * 4 + wave;        // 0..47, 64-wide column group
    const int b = blockIdx.z;                     // 0..3
    const int colBase = grp * 64;
    const int j = colBase / 768;                  // 768-wide block-diagonal block

    const int sg = sTile * 16 + llo;              // A-row for this lane
    const bf16_t* arow = Hb + ((size_t)(sg * 4 + b)) * 1024 + j * 256 + lhi * 16;
    const bf16_t* bcol = Wt + (size_t)(colBase + llo) * 1024 + b * 256 + lhi * 16;

    v8f acc[4];
#pragma unroll
    for (int c = 0; c < 4; ++c) acc[c] = (v8f){};

#pragma unroll
    for (int kk = 0; kk < 8; ++kk) {
        v8bf alo = *(const v8bf*)(arow + kk * 32);
        v8bf ahi = *(const v8bf*)(arow + kk * 32 + 8);
        v16bf a;
#pragma unroll
        for (int e = 0; e < 8; ++e) { a[e] = alo[e]; a[8 + e] = ahi[e]; }
#pragma unroll
        for (int c = 0; c < 4; ++c) {
            v16bf bm = *(const v16bf*)(bcol + (size_t)c * 16 * 1024 + kk * 32);
            acc[c] = __builtin_amdgcn_wmma_f32_16x16x32_bf16(false, a, false, bm,
                                                             (short)0, acc[c], false, false);
        }
    }

    const int g = colBase / 192;
    const int rcolBase = colBase % 192;           // 0 (Q), 64 (K) or 128 (V); wave-uniform
    if (rcolBase < 128) {                         // Q or K
#pragma unroll
        for (int c = 0; c < 4; ++c) {
            size_t base = ((size_t)b * 1024) * 2048 + g * 128 + rcolBase + c * 16 + llo;
#pragma unroll
            for (int e = 0; e < 8; ++e)
                QK[base + (size_t)(sTile * 16 + e + 8 * lhi) * 2048] = f2bf(acc[c][e]);
        }
    } else {                                      // V, transposed over t
#pragma unroll
        for (int c = 0; c < 4; ++c) {
            size_t vbase = ((size_t)(b * 16 + g) * 64 + c * 16 + llo) * 1024
                           + sTile * 16 + 8 * lhi;
#pragma unroll
            for (int e = 0; e < 8; ++e) Vt[vbase + e] = f2bf(acc[c][e]);
        }
    }
}

// ---------------- fused attention + polynorm ----------------
__global__ __launch_bounds__(128)
void attn(const bf16_t* __restrict__ QK, const bf16_t* __restrict__ Vt,
          const float* __restrict__ gamma, const float* __restrict__ beta,
          float* __restrict__ out) {
    __shared__ bf16_t lds[4][16 * 32];
    const int lane = threadIdx.x & 31;
    const int wave = threadIdx.x >> 5;
    const int lhi = lane >> 4, llo = lane & 15;
    const int sTile = blockIdx.x * 4 + wave;      // 0..63
    const int n = blockIdx.y;                     // head
    const int b = blockIdx.z;
    const float gm = gamma[n], bt = beta[n];
    const float norm = 0.125f;                    // 1/sqrt(64)

    // Q A-fragments (two k-steps covering h=0..63)
    v16bf qa[2];
    {
        const bf16_t* qrow = QK + ((size_t)(b * 1024) + sTile * 16 + llo) * 2048
                             + n * 128 + lhi * 16;
#pragma unroll
        for (int ks = 0; ks < 2; ++ks) {
            v8bf lo = *(const v8bf*)(qrow + ks * 32);
            v8bf hi = *(const v8bf*)(qrow + ks * 32 + 8);
#pragma unroll
            for (int e = 0; e < 8; ++e) { qa[ks][e] = lo[e]; qa[ks][8 + e] = hi[e]; }
        }
    }

    const bf16_t* kbase = QK + ((size_t)b * 1024) * 2048 + n * 128 + 64 + lhi * 16;
    const bf16_t* vbase = Vt + ((size_t)(b * 16 + n) * 64 + llo) * 1024 + lhi * 16;

    v8f O[4];  float rs[8];  float vs[4];
#pragma unroll
    for (int h = 0; h < 4; ++h) { O[h] = (v8f){}; vs[h] = 0.f; }
#pragma unroll
    for (int e = 0; e < 8; ++e) rs[e] = 0.f;

    for (int tB = 0; tB < 32; ++tB) {
        const int t0 = tB * 32;
        v8f s0 = {}, s1 = {};
#pragma unroll
        for (int ks = 0; ks < 2; ++ks) {
            v16bf k0 = *(const v16bf*)(kbase + (size_t)(t0 + llo) * 2048 + ks * 32);
            v16bf k1 = *(const v16bf*)(kbase + (size_t)(t0 + 16 + llo) * 2048 + ks * 32);
            s0 = __builtin_amdgcn_wmma_f32_16x16x32_bf16(false, qa[ks], false, k0,
                                                         (short)0, s0, false, false);
            s1 = __builtin_amdgcn_wmma_f32_16x16x32_bf16(false, qa[ks], false, k1,
                                                         (short)0, s1, false, false);
        }
        // polynomial sigmoid + stage probs into LDS (C-layout -> A-layout transpose)
#pragma unroll
        for (int e = 0; e < 8; ++e) {
            float a0 = polysig(s0[e] * norm);
            float a1 = polysig(s1[e] * norm);
            rs[e] += a0 + a1;
            int srow = e + 8 * lhi;
            lds[wave][srow * 32 + llo]      = f2bf(a0);
            lds[wave][srow * 32 + 16 + llo] = f2bf(a1);
        }
        __builtin_amdgcn_wave_barrier();
        asm volatile("s_wait_dscnt 0" ::: "memory");
        v16bf pa;
        {
            const bf16_t* lp = &lds[wave][llo * 32 + 16 * lhi];
#pragma unroll
            for (int e = 0; e < 8; ++e) { pa[e] = lp[e]; pa[8 + e] = lp[8 + e]; }
        }
        __builtin_amdgcn_wave_barrier();
        // probs @ V over this 32-wide t-block, 4 h-tiles; fold in colsum(V) for beta
#pragma unroll
        for (int ht = 0; ht < 4; ++ht) {
            v16bf vb = *(const v16bf*)(vbase + (size_t)ht * 16 * 1024 + t0);
            float p = 0.f;
#pragma unroll
            for (int e = 0; e < 16; ++e) p += bf2f(vb[e]);
            vs[ht] += p;
            O[ht] = __builtin_amdgcn_wmma_f32_16x16x32_bf16(false, pa, false, vb,
                                                            (short)0, O[ht], false, false);
        }
    }

    // row-sums: reduce across the 16 lanes sharing each M-group
#pragma unroll
    for (int e = 0; e < 8; ++e) {
        float r = rs[e];
        r += __shfl_xor(r, 1, 16);
        r += __shfl_xor(r, 2, 16);
        r += __shfl_xor(r, 4, 16);
        r += __shfl_xor(r, 8, 16);
        rs[e] = gm / (r + 1e-7f);
    }
    // V column sums: combine the two lane halves (same h = lane&15)
#pragma unroll
    for (int ht = 0; ht < 4; ++ht) vs[ht] += __shfl_xor(vs[ht], 16, 32);

#pragma unroll
    for (int ht = 0; ht < 4; ++ht) {
        int col = n * 64 + ht * 16 + llo;
        float bterm = bt * vs[ht];
#pragma unroll
        for (int e = 0; e < 8; ++e) {
            int srow = sTile * 16 + e + 8 * lhi;
            out[((size_t)(srow * 4 + b)) * 1024 + col] = O[ht][e] * rs[e] + bterm;
        }
    }
}

extern "C" void kernel_launch(void* const* d_in, const int* in_sizes, int n_in,
                              void* d_out, int out_size, void* d_ws, size_t ws_size,
                              hipStream_t stream) {
    const float* H     = (const float*)d_in[0];   // [1024,4,1024]
    const float* W     = (const float*)d_in[1];   // [1024,3072]
    const float* gamma = (const float*)d_in[2];   // [16]
    const float* beta  = (const float*)d_in[3];   // [16]
    float* out = (float*)d_out;

    char* ws = (char*)d_ws;
    bf16_t* Hb = (bf16_t*)(ws);                        //  8 MiB
    bf16_t* Wt = (bf16_t*)(ws + (size_t)(8  << 20));   //  6 MiB
    bf16_t* QK = (bf16_t*)(ws + (size_t)(14 << 20));   // 16 MiB [b][s][2048]
    bf16_t* Vt = (bf16_t*)(ws + (size_t)(30 << 20));   //  8 MiB [b][head][64][1024]

    const int nH = 1024 * 4 * 1024;
    const int nW = 1024 * 3072;
    cvt_hidden<<<(nH + 255) / 256, 256, 0, stream>>>(H, Hb, nH);
    cvt_weight_t<<<(nW + 255) / 256, 256, 0, stream>>>(W, Wt);
    qkv_gemm<<<dim3(64, 12, 4), 128, 0, stream>>>(Hb, Wt, QK, Vt);
    attn<<<dim3(16, 16, 4), 128, 0, stream>>>(QK, Vt, gamma, beta, out);
}
